// DecoderRNN_35631048688037
// MI455X (gfx1250) — matrix operand order repaired and checked by reading, not compiled, not run on hardware
//
#include <hip/hip_runtime.h>
#include <math.h>

// ---------------------------------------------------------------------------
// DecoderRNN (3-layer LSTM, H=1024, B=256, 128 steps) for gfx1250 (MI455X)
//  - bf16 weights/activations, f32 accumulate via v_wmma_f32_16x16x32_bf16
//  - layer-0 input GEMM hoisted out of the time loop (input is constant)
//  - weights pre-transposed to [N][K] so B-fragments are contiguous 32B loads
//  - 2x M register blocking per wave: each weight fragment feeds 2 WMMAs
//  - h double-buffered across steps; c updated in place (exact per-elem RMW)
// ---------------------------------------------------------------------------

typedef __attribute__((ext_vector_type(16))) __bf16 v16bf;
typedef __attribute__((ext_vector_type(8)))  __bf16 v8bf;
typedef __attribute__((ext_vector_type(8)))  float  v8f;

#define Bsz   256
#define Hh    1024
#define G4    4096          // 4*H
#define OUTD  204
#define OUTP  208           // 204 padded to 16
#define LEN   128
#define BH    (Bsz*Hh)

__device__ __forceinline__ unsigned short f2bf(float x) {
  union { float f; unsigned u; } v; v.f = x;
  unsigned r = v.u + 0x7FFFu + ((v.u >> 16) & 1u);   // round-to-nearest-even
  return (unsigned short)(r >> 16);
}
__device__ __forceinline__ float sigm(float x) { return 1.0f / (1.0f + __expf(-x)); }

__device__ __forceinline__ v16bf afrag(const unsigned short* p) {
  return *(const v16bf*)(p);          // K {0..15} (lanes 0-15) / {16..31} (lanes 16-31)
}

// One K=1024 GEMM pass: 2 M-fragments x 4 gate tiles per wave.
// X  : [256][1024] bf16 row-major (A rows = batch)
// Wt : [4096][1024] bf16 (pre-transposed: Wt[n][k] = W[k][n])
// acc[mi][g] accumulates rows {row0+mi*16 ..} x cols {g*1024 + col tile}
__device__ __forceinline__ void gemm8(v8f acc[2][4],
                                      const unsigned short* __restrict__ X,
                                      const unsigned short* __restrict__ Wt,
                                      int row0, int col, int c0, int kb) {
  const unsigned short* xr0 = X + (size_t)row0 * Hh;
  const unsigned short* xr1 = xr0 + (size_t)16 * Hh;
  const unsigned short* w0 = Wt + ((size_t)(0 * Hh + col)) * Hh + kb;
  const unsigned short* w1 = Wt + ((size_t)(1 * Hh + col)) * Hh + kb;
  const unsigned short* w2 = Wt + ((size_t)(2 * Hh + col)) * Hh + kb;
  const unsigned short* w3 = Wt + ((size_t)(3 * Hh + col)) * Hh + kb;
#pragma unroll 2
  for (int ks = 0; ks < Hh; ks += 32) {
    v16bf a0 = afrag(xr0 + ks + c0);
    v16bf a1 = afrag(xr1 + ks + c0);
    v16bf b0 = *(const v16bf*)(w0 + ks);
    v16bf b1 = *(const v16bf*)(w1 + ks);
    v16bf b2 = *(const v16bf*)(w2 + ks);
    v16bf b3 = *(const v16bf*)(w3 + ks);
    acc[0][0] = __builtin_amdgcn_wmma_f32_16x16x32_bf16(false, a0, false, b0, (short)0, acc[0][0], false, false);
    acc[1][0] = __builtin_amdgcn_wmma_f32_16x16x32_bf16(false, a1, false, b0, (short)0, acc[1][0], false, false);
    acc[0][1] = __builtin_amdgcn_wmma_f32_16x16x32_bf16(false, a0, false, b1, (short)0, acc[0][1], false, false);
    acc[1][1] = __builtin_amdgcn_wmma_f32_16x16x32_bf16(false, a1, false, b1, (short)0, acc[1][1], false, false);
    acc[0][2] = __builtin_amdgcn_wmma_f32_16x16x32_bf16(false, a0, false, b2, (short)0, acc[0][2], false, false);
    acc[1][2] = __builtin_amdgcn_wmma_f32_16x16x32_bf16(false, a1, false, b2, (short)0, acc[1][2], false, false);
    acc[0][3] = __builtin_amdgcn_wmma_f32_16x16x32_bf16(false, a0, false, b3, (short)0, acc[0][3], false, false);
    acc[1][3] = __builtin_amdgcn_wmma_f32_16x16x32_bf16(false, a1, false, b3, (short)0, acc[1][3], false, false);
  }
}

// One LSTM layer for one timestep.
// Xin/Wt_ih may be null (layer 0: input contribution precomputed in xpre).
// xpre: [256][4096] f32 (layer0, biases baked in) OR bias: [4096] f32 (layers 1,2)
// Grid: 64 blocks x 256 threads. Block b: M-group (b>>3)*32, N-tiles (b&7)*8 + wave.
__global__ void __launch_bounds__(256) k_lstm_layer(
    const unsigned short* __restrict__ Xin,  const unsigned short* __restrict__ Wt_ih,
    const unsigned short* __restrict__ Hprev,const unsigned short* __restrict__ Wt_hh,
    const float* __restrict__ xpre, const float* __restrict__ bias,
    float* __restrict__ C, unsigned short* __restrict__ Hnext) {
  const int lane = threadIdx.x & 31;
  const int w    = threadIdx.x >> 5;
  const int mt2  = blockIdx.x >> 3;                 // 0..7   (M groups of 32 rows)
  const int nt   = (blockIdx.x & 7) * 8 + w;        // 0..63  (N tiles of 16)
  const int row0 = mt2 * 32 + (lane & 15);
  const int col  = nt * 16 + (lane & 15);
  const int c0   = (lane >> 4) * 16;
  const int kb   = (lane >> 4) * 16;

  v8f acc[2][4] = {};
  if (Xin) gemm8(acc, Xin, Wt_ih, row0, col, c0, kb);     // x @ W_ih (layers 1,2)
  gemm8(acc, Hprev, Wt_hh, row0, col, c0, kb);            // h @ W_hh

#pragma unroll
  for (int mi = 0; mi < 2; ++mi) {
    const int rbase = mt2 * 32 + mi * 16 + (lane >> 4) * 8;   // C/D layout rows
#pragma unroll
    for (int e = 0; e < 8; ++e) {
      const int r = rbase + e;
      float gi = acc[mi][0][e], gf = acc[mi][1][e], gg = acc[mi][2][e], go = acc[mi][3][e];
      if (xpre) {
        const float* xp = xpre + (size_t)r * G4 + col;
        gi += xp[0]; gf += xp[Hh]; gg += xp[2 * Hh]; go += xp[3 * Hh];
      } else {
        gi += bias[col]; gf += bias[Hh + col]; gg += bias[2 * Hh + col]; go += bias[3 * Hh + col];
      }
      const float iv = sigm(gi), fv = sigm(gf), gv = tanhf(gg), ov = sigm(go);
      const size_t idx = (size_t)r * Hh + col;
      const float cn = fv * C[idx] + iv * gv;
      C[idx] = cn;
      Hnext[idx] = f2bf(ov * tanhf(cn));
    }
  }
}

// FC head: out[b, t, :204] = h2 @ fc_W + fc_b   (Wt: [208][1024] bf16, padded)
__global__ void __launch_bounds__(256) k_fc(const unsigned short* __restrict__ H,
                                            const unsigned short* __restrict__ Wt,
                                            const float* __restrict__ fcb,
                                            float* __restrict__ out, int t) {
  const int lane = threadIdx.x & 31;
  const int tile = blockIdx.x * 8 + (threadIdx.x >> 5);
  if (tile >= 16 * 13) return;                            // whole-wave exit (EXEC stays full)
  const int mt = tile / 13, nt = tile % 13;
  const int row = mt * 16 + (lane & 15);
  const int col = nt * 16 + (lane & 15);
  const int c0 = (lane >> 4) * 16, kb = (lane >> 4) * 16;
  v8f acc = {};
  const unsigned short* xr = H + (size_t)row * Hh;
  const unsigned short* wp = Wt + (size_t)col * Hh + kb;
#pragma unroll 2
  for (int ks = 0; ks < Hh; ks += 32) {
    v16bf a = afrag(xr + ks + c0);
    v16bf b = *(const v16bf*)(wp + ks);
    acc = __builtin_amdgcn_wmma_f32_16x16x32_bf16(false, a, false, b, (short)0, acc, false, false);
  }
  if (col < OUTD) {
    const float bv = fcb[col];
    const int rbase = mt * 16 + (lane >> 4) * 8;
#pragma unroll
    for (int e = 0; e < 8; ++e)
      out[(size_t)(rbase + e) * (LEN * OUTD) + (size_t)t * OUTD + col] = acc[e] + bv;
  }
}

// ---------------- one-time setup kernels ----------------

// Wt[n][k] = bf16(W[k][n]); zero-pad cols n >= N.
__global__ void k_transpose(const float* __restrict__ W, unsigned short* __restrict__ Wt,
                            int K, int N, int Npad) {
  const int idx = blockIdx.x * 256 + threadIdx.x;
  if (idx >= Npad * K) return;
  const int n = idx / K, k = idx - n * K;
  Wt[idx] = f2bf((n < N) ? W[(size_t)k * N + n] : 0.0f);
}

__global__ void k_bias2(const float* b1a, const float* b1b,
                        const float* b2a, const float* b2b, float* bias) {
  const int i = blockIdx.x * 256 + threadIdx.x;
  if (i < G4)            bias[i] = b1a[i] + b1b[i];
  else if (i < 2 * G4)   { const int j = i - G4; bias[i] = b2a[j] + b2b[j]; }
}

// h0/c0 init: frame @ inh_W + inh_b (and inc_*), split into 3 layers.
__global__ void k_init_state(const float* __restrict__ frame,
                             const float* __restrict__ inhW, const float* __restrict__ inhb,
                             const float* __restrict__ incW, const float* __restrict__ incb,
                             unsigned short* __restrict__ Hbuf, float* __restrict__ Cbuf) {
  const int idx = blockIdx.x * 256 + threadIdx.x;
  if (idx >= Bsz * 3 * Hh) return;
  const int b = idx / (3 * Hh), j = idx % (3 * Hh);
  const int l = j / Hh, jj = j - l * Hh;
  const float* fr = frame + (size_t)b * OUTD;
  float hv = inhb[j], cv = incb[j];
  for (int k = 0; k < OUTD; ++k) {
    const float x = fr[k];
    hv += x * inhW[(size_t)k * (3 * Hh) + j];
    cv += x * incW[(size_t)k * (3 * Hh) + j];
  }
  Hbuf[(size_t)(l * 2) * BH + (size_t)b * Hh + jj] = f2bf(hv);  // parity-0 buffer
  Cbuf[(size_t)l * BH + (size_t)b * Hh + jj] = cv;
}

// Time-invariant layer-0 input contribution: x_in @ W_ih0 + b_ih0 + b_hh0.
__global__ void k_xpre(const float* __restrict__ frame, const int* __restrict__ labels,
                       const float* __restrict__ Wih0, const float* __restrict__ bih0,
                       const float* __restrict__ bhh0, float* __restrict__ xpre) {
  const int idx = blockIdx.x * 256 + threadIdx.x;
  if (idx >= Bsz * G4) return;
  const int b = idx / G4, n = idx - b * G4;
  float s = bih0[n] + bhh0[n];
  const float* fr = frame + (size_t)b * OUTD;
  for (int k = 0; k < OUTD; ++k) s += fr[k] * Wih0[(size_t)k * G4 + n];
  s += Wih0[(size_t)(OUTD + labels[0]) * G4 + n];   // one-hot contribution
  xpre[idx] = s;
}

// ---------------------------------------------------------------------------

extern "C" void kernel_launch(void* const* d_in, const int* in_sizes, int n_in,
                              void* d_out, int out_size, void* d_ws, size_t ws_size,
                              hipStream_t stream) {
  const float* inputs = (const float*)d_in[0];
  const float* W_ih0 = (const float*)d_in[1];
  const float* W_hh0 = (const float*)d_in[2];
  const float* b_ih0 = (const float*)d_in[3];
  const float* b_hh0 = (const float*)d_in[4];
  const float* W_ih1 = (const float*)d_in[5];
  const float* W_hh1 = (const float*)d_in[6];
  const float* b_ih1 = (const float*)d_in[7];
  const float* b_hh1 = (const float*)d_in[8];
  const float* W_ih2 = (const float*)d_in[9];
  const float* W_hh2 = (const float*)d_in[10];
  const float* b_ih2 = (const float*)d_in[11];
  const float* b_hh2 = (const float*)d_in[12];
  const float* fc_W  = (const float*)d_in[13];
  const float* fc_b  = (const float*)d_in[14];
  const float* inh_W = (const float*)d_in[15];
  const float* inh_b = (const float*)d_in[16];
  const float* inc_W = (const float*)d_in[17];
  const float* inc_b = (const float*)d_in[18];
  const int*   labels= (const int*)d_in[19];
  float* out = (float*)d_out;

  char* base = (char*)d_ws;
  size_t off = 0;
  auto alloc = [&](size_t bytes) -> void* {
    void* p = base + off;
    off = (off + bytes + 255) & ~(size_t)255;
    return p;
  };
  unsigned short* WtHH0 = (unsigned short*)alloc((size_t)G4 * Hh * 2);
  unsigned short* WtHH1 = (unsigned short*)alloc((size_t)G4 * Hh * 2);
  unsigned short* WtHH2 = (unsigned short*)alloc((size_t)G4 * Hh * 2);
  unsigned short* WtIH1 = (unsigned short*)alloc((size_t)G4 * Hh * 2);
  unsigned short* WtIH2 = (unsigned short*)alloc((size_t)G4 * Hh * 2);
  unsigned short* WtFC  = (unsigned short*)alloc((size_t)OUTP * Hh * 2);
  float*          xpre  = (float*)alloc((size_t)Bsz * G4 * 4);
  float*          bias12= (float*)alloc((size_t)2 * G4 * 4);
  unsigned short* Hbuf  = (unsigned short*)alloc((size_t)3 * 2 * BH * 2);  // [layer][parity]
  float*          Cbuf  = (float*)alloc((size_t)3 * BH * 4);

  // ---- one-time setup (recomputed every call: deterministic) ----
  {
    const int nW = G4 * Hh, gW = (nW + 255) / 256;
    k_transpose<<<gW, 256, 0, stream>>>(W_hh0, WtHH0, Hh, G4, G4);
    k_transpose<<<gW, 256, 0, stream>>>(W_hh1, WtHH1, Hh, G4, G4);
    k_transpose<<<gW, 256, 0, stream>>>(W_hh2, WtHH2, Hh, G4, G4);
    k_transpose<<<gW, 256, 0, stream>>>(W_ih1, WtIH1, Hh, G4, G4);
    k_transpose<<<gW, 256, 0, stream>>>(W_ih2, WtIH2, Hh, G4, G4);
    const int nF = OUTP * Hh;
    k_transpose<<<(nF + 255) / 256, 256, 0, stream>>>(fc_W, WtFC, Hh, OUTD, OUTP);
    k_bias2<<<(2 * G4 + 255) / 256, 256, 0, stream>>>(b_ih1, b_hh1, b_ih2, b_hh2, bias12);
    k_init_state<<<(Bsz * 3 * Hh + 255) / 256, 256, 0, stream>>>(
        inputs, inh_W, inh_b, inc_W, inc_b, Hbuf, Cbuf);
    k_xpre<<<(Bsz * G4 + 255) / 256, 256, 0, stream>>>(
        inputs, labels, W_ih0, b_ih0, b_hh0, xpre);
  }

  // ---- 128 sequential steps: L0, L1, L2, FC per step ----
  for (int t = 0; t < LEN; ++t) {
    const int p = t & 1;
    unsigned short* h0r = Hbuf + (size_t)(0 * 2 + p)       * BH;
    unsigned short* h0w = Hbuf + (size_t)(0 * 2 + (1 - p)) * BH;
    unsigned short* h1r = Hbuf + (size_t)(1 * 2 + p)       * BH;
    unsigned short* h1w = Hbuf + (size_t)(1 * 2 + (1 - p)) * BH;
    unsigned short* h2r = Hbuf + (size_t)(2 * 2 + p)       * BH;
    unsigned short* h2w = Hbuf + (size_t)(2 * 2 + (1 - p)) * BH;

    k_lstm_layer<<<64, 256, 0, stream>>>(nullptr, nullptr, h0r, WtHH0,
                                         xpre, nullptr, Cbuf + 0 * (size_t)BH, h0w);
    k_lstm_layer<<<64, 256, 0, stream>>>(h0w, WtIH1, h1r, WtHH1,
                                         nullptr, bias12, Cbuf + 1 * (size_t)BH, h1w);
    k_lstm_layer<<<64, 256, 0, stream>>>(h1w, WtIH2, h2r, WtHH2,
                                         nullptr, bias12 + G4, Cbuf + 2 * (size_t)BH, h2w);
    k_fc<<<26, 256, 0, stream>>>(h2w, WtFC, fc_b, out, t);
  }
}
